// Head_67877663146711
// MI455X (gfx1250) — matrix operand order repaired and checked by reading, not compiled, not run on hardware
//
#include <hip/hip_runtime.h>

// ---------------------------------------------------------------------------
// Causal single-head attention for MI455X (gfx1250, wave32, WMMA).
//   B=4, S=4096, E=1024, H=64
// Path: fp32 inputs -> bf16 operands, v_wmma_f32_16x16x32_bf16, fp32 softmax.
// Attention uses a TRANSPOSED score layout (S^T = K @ Q^T) so the softmax
// reduction over keys is in-lane (1 shfl_xor per reduction instead of 4x16),
// and O^T = V^T @ P^T so the pre-transposed V feeds the A operand directly.
// The 1/sqrt(H) softmax scale is folded into Q at projection time.
// ---------------------------------------------------------------------------

#define B_ 4
#define S_ 4096
#define E_ 1024
#define H_ 64

typedef __bf16 bf16;
typedef __attribute__((ext_vector_type(16))) __bf16 v16bf;
typedef __attribute__((ext_vector_type(8)))  __bf16 v8bf;
typedef __attribute__((ext_vector_type(8)))  float  v8f;
typedef __attribute__((ext_vector_type(8)))  float  f32x8;

union ABf { v16bf v; v8bf h[2]; };

static __device__ __forceinline__ v8f zero8() {
  v8f z = {0.f,0.f,0.f,0.f,0.f,0.f,0.f,0.f};
  return z;
}

// D = A(16x32 bf16) * B(32x16 bf16) + C(16x16 f32)
static __device__ __forceinline__ v8f wmma_bf16(v16bf a, v16bf b, v8f c) {
  return __builtin_amdgcn_wmma_f32_16x16x32_bf16(false, a, false, b,
                                                 (short)0, c, false, false);
}

// ---------------------------------------------------------------------------
// Kernel 1: W[E][H] fp32  ->  Wt[H][E] bf16 (so WMMA B-fragments are
// contiguous 16-bf16 runs per lane).
// ---------------------------------------------------------------------------
__global__ void prep_weights(const float* __restrict__ Wq,
                             const float* __restrict__ Wk,
                             const float* __restrict__ Wv,
                             bf16* __restrict__ Wtq,
                             bf16* __restrict__ Wtk,
                             bf16* __restrict__ Wtv) {
  int idx = blockIdx.x * blockDim.x + threadIdx.x;   // exactly 3*E*H threads
  const int per = E_ * H_;
  int mat = idx / per;
  int r   = idx % per;
  int h   = r / E_;          // output-coalesced along e
  int e   = r % E_;
  const float* W  = (mat == 0) ? Wq  : (mat == 1) ? Wk  : Wv;
  bf16*        Wt = (mat == 0) ? Wtq : (mat == 1) ? Wtk : Wtv;
  Wt[(size_t)h * E_ + e] = (bf16)W[(size_t)e * H_ + h];
}

// ---------------------------------------------------------------------------
// Kernel 2: QKV projection. One wave computes 16 rows x 64 cols for each of
// Q, K, V using v_wmma_f32_16x16x32_bf16, K-loop over E in steps of 32.
//
// 16-bit A layout (ISA 7.12.2): lane L (g=L/16, m=L%16) holds row m,
//   elements 0..7  = K in [8g   .. 8g+7]
//   elements 8..15 = K in [16+8g.. 16+8g+7]
// B here (and A, same order): lane L (g, n=L%16) holds column n, K in [16g .. 16g+15].
// C/D layout: lane L (g, n) VGPR r = out[row 8g+r][n].
//
// Q stored PRE-SCALED by 1/sqrt(H); Q,K row-major bf16 [B*S][H];
// V stored transposed bf16 [B][H][S].
// ---------------------------------------------------------------------------
__global__ void __launch_bounds__(128)
qkv_proj(const float* __restrict__ x,
         const bf16* __restrict__ Wtq, const bf16* __restrict__ Wtk,
         const bf16* __restrict__ Wtv,
         const float* __restrict__ bq, const float* __restrict__ bk,
         const float* __restrict__ bv,
         bf16* __restrict__ Qb, bf16* __restrict__ Kb, bf16* __restrict__ Vt) {
  const int lane = threadIdx.x & 31;
  const int wid  = (blockIdx.x * blockDim.x + threadIdx.x) >> 5;
  const int row0 = wid * 16;                 // 16-row tile, never crosses batch
  const int g  = lane >> 4;
  const int nl = lane & 15;

  for (int which = 0; which < 3; ++which) {
    const bf16*  Wt   = (which == 0) ? Wtq : (which == 1) ? Wtk : Wtv;
    const float* bias = (which == 0) ? bq  : (which == 1) ? bk  : bv;
    const float  post = (which == 0) ? 0.125f : 1.0f;  // fold 1/sqrt(64) into Q

    v8f acc[4];
    #pragma unroll
    for (int ng = 0; ng < 4; ++ng) acc[ng] = zero8();

    for (int kc = 0; kc < E_; kc += 32) {
      // A fragment: x[row0+nl][kc + 16g .. 16g+15]
      ABf a;
      const float* xr = x + (size_t)(row0 + nl) * E_ + kc;
      f32x8 xl0 = *(const f32x8*)(xr + 16 * g);
      f32x8 xl1 = *(const f32x8*)(xr + 16 * g + 8);
      #pragma unroll
      for (int i = 0; i < 8; ++i) {
        a.h[0][i] = (bf16)xl0[i];
        a.h[1][i] = (bf16)xl1[i];
      }
      #pragma unroll
      for (int ng = 0; ng < 4; ++ng) {
        const int col = 16 * ng + nl;
        v16bf bm = *(const v16bf*)(Wt + (size_t)col * E_ + kc + 16 * g);
        acc[ng] = wmma_bf16(a.v, bm, acc[ng]);
      }
    }

    if (which < 2) {                               // Q, K: row-major bf16
      bf16* Out = (which == 0) ? Qb : Kb;
      #pragma unroll
      for (int ng = 0; ng < 4; ++ng) {
        const float bb = bias[16 * ng + nl];
        #pragma unroll
        for (int r = 0; r < 8; ++r)
          Out[(size_t)(row0 + 8 * g + r) * H_ + 16 * ng + nl] =
              (bf16)((acc[ng][r] + bb) * post);
      }
    } else {                                       // V: transposed [B][H][S]
      const int b    = row0 / S_;
      const int sloc = row0 % S_;
      #pragma unroll
      for (int ng = 0; ng < 4; ++ng) {
        const int hd = 16 * ng + nl;
        const float bb = bias[hd];
        v8bf pk;
        #pragma unroll
        for (int r = 0; r < 8; ++r) pk[r] = (bf16)(acc[ng][r] + bb);
        *(v8bf*)(Vt + ((size_t)b * H_ + hd) * S_ + sloc + 8 * g) = pk;
      }
    }
  }
}

// ---------------------------------------------------------------------------
// Kernel 3: flash attention, transposed layout. One wave owns 16 queries
// (queries live in lanes, keys live in per-lane VGPRs):
//
//   S^T(32k x 16q) = K(32k x 64hd) @ Qs^T(64hd x 16q)   (Q pre-scaled)
//     A = K rows (contiguous in Kb), B = Q^T (contiguous 32B runs, loaded
//     once per wave, loop-invariant).
//   softmax over keys: in-lane reduce over 16 VGPR values + one shfl_xor(16)
//     (lane pair n / n+16 holds complementary key subsets).
//   O^T(64hd x 16q) += V^T(64hd x 32k) @ P^T(32k x 16q)
//     A = V^T rows (contiguous in pre-transposed Vt), B = P^T from a 1KB
//     per-wave LDS round-trip (2x16B DS stores + 1x32B DS load per block;
//     same-wave DS ops are architecturally in-order).
//
// Epilogue stores O^T as 4x32B contiguous fp32 vector stores per lane.
// ---------------------------------------------------------------------------
__global__ void __launch_bounds__(128)
attn_fwd(const bf16* __restrict__ Qb, const bf16* __restrict__ Kb,
         const bf16* __restrict__ Vt, float* __restrict__ out) {
  __shared__ bf16 Pl[4][16 * 32];                  // 4 waves/block x 1KB

  const int lane   = threadIdx.x & 31;
  const int wlocal = threadIdx.x >> 5;
  const int wid    = (blockIdx.x * blockDim.x + threadIdx.x) >> 5;
  const int b  = wid / (S_ / 16);
  const int q0 = (wid % (S_ / 16)) * 16;
  const int g  = lane >> 4;
  const int nl = lane & 15;                        // query index within tile

  const bf16* Qbase = Qb + (size_t)b * S_ * H_;
  const bf16* Kbase = Kb + (size_t)b * S_ * H_;
  const bf16* Vbase = Vt + (size_t)b * H_ * S_;

  // Q^T as two loop-invariant B fragments (head-dim chunks [0,32),[32,64)):
  // lane (g,n): Q[q0+n][32c + 16g .. 16g+15], one 32B load each.
  v16bf qb0, qb1;
  {
    const bf16* qr = Qbase + (size_t)(q0 + nl) * H_;
    qb0 = *(const v16bf*)(qr + 16 * g);
    qb1 = *(const v16bf*)(qr + 32 + 16 * g);
  }

  float mi = -3.0e38f, li = 0.f;                   // per-query (per-lane)
  v8f acc[4];                                      // O^T: acc[hg] r -> hd 16hg+8g+r
  #pragma unroll
  for (int hg = 0; hg < 4; ++hg) acc[hg] = zero8();

  for (int kb = 0; kb <= q0 + 15; kb += 32) {
    // ---- S^T tiles: two 16-key groups; keys = 16kg + 8g + r ----
    v8f st[2];
    #pragma unroll
    for (int kg = 0; kg < 2; ++kg) {
      const bf16* kr = Kbase + (size_t)(kb + 16 * kg + nl) * H_;
      ABf ka0, ka1;                                // K rows as A fragments
      ka0.h[0] = *(const v8bf*)(kr + 16 * g);
      ka0.h[1] = *(const v8bf*)(kr + 8 + 16 * g);
      ka1.h[0] = *(const v8bf*)(kr + 32 + 16 * g);
      ka1.h[1] = *(const v8bf*)(kr + 40 + 16 * g);
      v8f t  = wmma_bf16(ka0.v, qb0, zero8());
      st[kg] = wmma_bf16(ka1.v, qb1, t);
    }

    const bool full = (kb + 31 <= q0);             // block fully below diagonal
    if (!full) {
      #pragma unroll
      for (int kg = 0; kg < 2; ++kg)
        #pragma unroll
        for (int r = 0; r < 8; ++r) {
          const int key = kb + 16 * kg + 8 * g + r;
          if (key > q0 + nl) st[kg][r] = -3.0e38f;
        }
    }

    // ---- online softmax: in-lane reduce + one cross-half shuffle ----
    float bm = st[0][0];
    #pragma unroll
    for (int r = 1; r < 8; ++r) bm = fmaxf(bm, st[0][r]);
    #pragma unroll
    for (int r = 0; r < 8; ++r) bm = fmaxf(bm, st[1][r]);
    bm = fmaxf(bm, __shfl_xor(bm, 16, 32));

    const float mnew  = fmaxf(mi, bm);
    const float alpha = __expf(mi - mnew);
    float p[2][8];
    float rs = 0.f;
    #pragma unroll
    for (int kg = 0; kg < 2; ++kg)
      #pragma unroll
      for (int r = 0; r < 8; ++r) {
        p[kg][r] = __expf(st[kg][r] - mnew);
        rs += p[kg][r];
      }
    rs += __shfl_xor(rs, 16, 32);
    li = li * alpha + rs;
    mi = mnew;
    #pragma unroll
    for (int hg = 0; hg < 4; ++hg)
      #pragma unroll
      for (int r = 0; r < 8; ++r) acc[hg][r] *= alpha;

    // ---- P^T -> B layout via per-wave LDS [query][key] = [16][32] bf16 ----
    bf16* P = &Pl[wlocal][0];
    #pragma unroll
    for (int kg = 0; kg < 2; ++kg) {
      v8bf pk;
      #pragma unroll
      for (int r = 0; r < 8; ++r) pk[r] = (bf16)p[kg][r];
      *(v8bf*)(P + nl * 32 + 16 * kg + 8 * g) = pk;   // keys contiguous
    }
    __builtin_amdgcn_wave_barrier();               // compiler ordering fence
    // B fragment: lane (g,n) holds query n, keys [16g .. 16g+15]
    v16bf pb = *(const v16bf*)(P + nl * 32 + 16 * g);
    __builtin_amdgcn_wave_barrier();

    // ---- O^T += V^T @ P^T: A = V^T rows (contiguous in Vt) ----
    #pragma unroll
    for (int hg = 0; hg < 4; ++hg) {
      const bf16* vr = Vbase + (size_t)(16 * hg + nl) * S_ + kb;
      ABf va;
      va.h[0] = *(const v8bf*)(vr + 16 * g);
      va.h[1] = *(const v8bf*)(vr + 8 + 16 * g);
      acc[hg] = wmma_bf16(va.v, pb, acc[hg]);
    }
  }

  // ---- epilogue: O^T C-layout -> contiguous 32B fp32 stores ----
  // lane (g,n), acc[hg][r] = O[q0+n][16hg + 8g + r]
  const float inv = 1.0f / li;
  float* orow = out + ((size_t)b * S_ + q0 + nl) * H_;
  #pragma unroll
  for (int hg = 0; hg < 4; ++hg) {
    f32x8 o;
    #pragma unroll
    for (int r = 0; r < 8; ++r) o[r] = acc[hg][r] * inv;
    *(f32x8*)(orow + 16 * hg + 8 * g) = o;
  }
}

// ---------------------------------------------------------------------------
// Host launcher
// ---------------------------------------------------------------------------
extern "C" void kernel_launch(void* const* d_in, const int* in_sizes, int n_in,
                              void* d_out, int out_size, void* d_ws,
                              size_t ws_size, hipStream_t stream) {
  (void)in_sizes; (void)n_in; (void)out_size; (void)ws_size;

  const float* x  = (const float*)d_in[0];
  const float* Wq = (const float*)d_in[1];
  const float* bq = (const float*)d_in[2];
  const float* Wk = (const float*)d_in[3];
  const float* bk = (const float*)d_in[4];
  const float* Wv = (const float*)d_in[5];
  const float* bv = (const float*)d_in[6];
  float* out = (float*)d_out;

  // Workspace layout (bf16): Qb, Kb [B*S][H]; Vt [B][H][S]; 3x Wt [H][E]
  bf16* Qb  = (bf16*)d_ws;
  bf16* Kb  = Qb  + (size_t)B_ * S_ * H_;
  bf16* Vt  = Kb  + (size_t)B_ * S_ * H_;
  bf16* Wtq = Vt  + (size_t)B_ * S_ * H_;
  bf16* Wtk = Wtq + (size_t)E_ * H_;
  bf16* Wtv = Wtk + (size_t)E_ * H_;

  prep_weights<<<(3 * E_ * H_) / 256, 256, 0, stream>>>(Wq, Wk, Wv,
                                                        Wtq, Wtk, Wtv);
  qkv_proj<<<(B_ * S_ / 16) / 4, 128, 0, stream>>>(x, Wtq, Wtk, Wtv,
                                                   bq, bk, bv, Qb, Kb, Vt);
  attn_fwd<<<(B_ * S_ / 16) / 4, 128, 0, stream>>>(Qb, Kb, Vt, out);
}
